// KNRM_24008867184958
// MI455X (gfx1250) — matrix operand order repaired and checked by reading, not compiled, not run on hardware
//
#include <hip/hip_runtime.h>

typedef __attribute__((ext_vector_type(16))) _Float16 v16h;
typedef __attribute__((ext_vector_type(4)))  _Float16 v4h;
typedef __attribute__((ext_vector_type(8)))  float    v8f;
typedef __attribute__((ext_vector_type(4)))  float    v4f;

#define VOCAB 50000
#define DIM   128
#define BATCH 1024
#define LQ    32
#define LD    512
#define KNUM  21

// exp(-50*mu_k^2) for the 20 shared-sigma bins, mu_k = -0.95 + 0.1k
__constant__ float KSCALE[20] = {
    2.5262e-20f, 2.0470e-16f, 6.1020e-13f, 6.6916e-10f, 2.69958e-7f,
    4.00653e-5f, 2.1874e-3f,  4.3937e-2f,  0.324652f,   0.882497f,
    0.882497f,   0.324652f,   4.3937e-2f,  2.1874e-3f,  4.00653e-5f,
    2.69958e-7f, 6.6916e-10f, 6.1020e-13f, 2.0470e-16f, 2.5262e-20f
};

#define C_Y  144.269504089f   // 100 * log2(e)
#define C_P  -72.1347520444f  // -50 * log2(e)
#define C_EX -721347.520444f  // -500000 * log2(e)

union Frag {
    v16h h;
    v4f  f[2];
};

// single-instruction cross-half exchange: ds_swizzle_b32 xor_mask=16
// (offset[15]=0 group-of-32 mode: and=0x1f, or=0, xor=0x10 -> 0x401f)
__device__ __forceinline__ float xor16(float x) {
    return __int_as_float(
        __builtin_amdgcn_ds_swizzle(__float_as_int(x), 0x401f));
}

// ---------------------------------------------------------------------------
// Kernel 1: L2-normalize the whole embedding table once, emit f16.
// ---------------------------------------------------------------------------
__global__ void __launch_bounds__(256)
normalize_emb_kernel(const float* __restrict__ emb, _Float16* __restrict__ nrm) {
    const int wave = threadIdx.x >> 5;
    const int lane = threadIdx.x & 31;
    const int row  = blockIdx.x * 8 + wave;
    if (row >= VOCAB) return;

    const v4f v = *(const v4f*)(emb + (size_t)row * DIM + lane * 4);
    float s = v.x * v.x + v.y * v.y + v.z * v.z + v.w * v.w;
#pragma unroll
    for (int m = 1; m < 32; m <<= 1) s += __shfl_xor(s, m, 32);

    const float scale = 1.0f / fmaxf(sqrtf(s), 1e-12f);
    v4h o;
    o.x = (_Float16)(v.x * scale);
    o.y = (_Float16)(v.y * scale);
    o.z = (_Float16)(v.z * scale);
    o.w = (_Float16)(v.w * scale);
    *(v4h*)(nrm + (size_t)row * DIM + lane * 4) = o;
}

// ---------------------------------------------------------------------------
// Kernel 2: one block per (batch element, pair). 8 waves.
// Wave w owns doc rows [w*64, w*64+64). DOCS are the A matrix (M), QUERIES
// are the B matrix (N): D-tile lane = query, VGPR index = doc row, so the
// sum over Ld stays in-register (+ one xor-16 exchange), no butterfly.
//   acc[dt][qt]: dt = 4 doc M-tiles, qt = 2 query N-tiles, K=128.
// ---------------------------------------------------------------------------
__global__ void __launch_bounds__(256)
knrm_kernel(const _Float16* __restrict__ nrm,
            const int* __restrict__ q1, const int* __restrict__ d1,
            const int* __restrict__ q2, const int* __restrict__ d2,
            const float* __restrict__ w0, const float* __restrict__ b0,
            const float* __restrict__ w1, const float* __restrict__ b1,
            const float* __restrict__ w2, const float* __restrict__ b2,
            float* __restrict__ logits) {
    const int b    = blockIdx.x;
    const int pair = blockIdx.y;
    const int* __restrict__ qp = pair ? q2 : q1;
    const int* __restrict__ dp = pair ? d2 : d1;

    const int  tid  = threadIdx.x;
    const int  wave = tid >> 5;
    const int  lane = tid & 31;
    const int  l15  = lane & 15;
    const bool hi   = lane >= 16;

    __shared__ float s_feat[LQ][KNUM + 1];   // [query][bin], padded rows
    __shared__ float s_f[KNUM];
    __shared__ float s_h0[10];
    __shared__ float s_h1[5];

    for (int i = tid; i < LQ * (KNUM + 1); i += 256) ((float*)s_feat)[i] = 0.0f;
    __syncthreads();

    // --- gather row indices ---------------------------------------------
    // A (docs, 16x32 tile): lane holds doc row dt*16 + (lane&15)
    // B (queries, 32x16 tile): lane holds query col qt*16 + (lane&15)
    int di[4], qi[2];
    const int n0 = wave * 64;
#pragma unroll
    for (int dt = 0; dt < 4; ++dt) di[dt] = dp[b * LD + n0 + dt * 16 + l15];
#pragma unroll
    for (int qt = 0; qt < 2; ++qt) qi[qt] = qp[b * LQ + qt * 16 + l15];

    const _Float16* aptr[4];
    const _Float16* bptr[2];
#pragma unroll
    for (int dt = 0; dt < 4; ++dt) aptr[dt] = nrm + (size_t)di[dt] * DIM;
#pragma unroll
    for (int qt = 0; qt < 2; ++qt) bptr[qt] = nrm + (size_t)qi[qt] * DIM;

    // ISA fragment layouts (16-bit A 16x32; B 32x16):
    //   A: lane<16 -> K base 0, lane>=16 -> K base 8; halves 0..7 => K+0..7,
    //      halves 8..15 => K+16..23  (two contiguous 16B runs)
    //   B: same as A: lane<16 -> K base 0, lane>=16 -> K base 8; K+0..7, K+16..23
    const int abase = hi ? 8 : 0;
    const int bbase = hi ? 8 : 0;

    v8f acc[4][2];
    const v8f vzero = {0.f, 0.f, 0.f, 0.f, 0.f, 0.f, 0.f, 0.f};
#pragma unroll
    for (int dt = 0; dt < 4; ++dt)
#pragma unroll
        for (int qt = 0; qt < 2; ++qt) acc[dt][qt] = vzero;

    // --- K loop: 128 = 4 x 32 -------------------------------------------
#pragma unroll
    for (int kk = 0; kk < DIM; kk += 32) {
        Frag a[4], bb[2];
#pragma unroll
        for (int dt = 0; dt < 4; ++dt) {
            a[dt].f[0] = *(const v4f*)(aptr[dt] + kk + abase);
            a[dt].f[1] = *(const v4f*)(aptr[dt] + kk + 16 + abase);
        }
#pragma unroll
        for (int qt = 0; qt < 2; ++qt) {
            bb[qt].f[0] = *(const v4f*)(bptr[qt] + kk + bbase);
            bb[qt].f[1] = *(const v4f*)(bptr[qt] + kk + 16 + bbase);
        }
#pragma unroll
        for (int dt = 0; dt < 4; ++dt)
#pragma unroll
            for (int qt = 0; qt < 2; ++qt)
                acc[dt][qt] = __builtin_amdgcn_wmma_f32_16x16x32_f16(
                    false, a[dt].h, false, bb[qt].h,
                    (short)0, acc[dt][qt], false, false);
    }

    // D-tile layout: lane&15 = query col, VGPR v = doc row v + (hi?8:0).
    // Per lane: sum over its 32 doc values; xor-16 partner holds the
    // complementary 32 docs of the SAME query -> one exchange finishes Ld.

    // --- exact-match bin (k=20): direct (x-1)^2 form for precision ------
#pragma unroll
    for (int qt = 0; qt < 2; ++qt) {
        float sv = 0.0f;
#pragma unroll
        for (int dt = 0; dt < 4; ++dt)
#pragma unroll
            for (int v = 0; v < 8; ++v) {
                const float t = acc[dt][qt][v] - 1.0f;
                sv += __builtin_amdgcn_exp2f(t * t * C_EX);
            }
        sv += xor16(sv);
        if (!hi) atomicAdd(&s_feat[qt * 16 + l15][20], sv);
    }

    // --- 20 soft bins, factorized:
    //  exp(-50(x-mu)^2) = exp(-50 mu^2) * 2^( fma(mu, 144.2695x, -72.13475x^2) )
    float yv[4][2][8], pv[4][2][8];
#pragma unroll
    for (int dt = 0; dt < 4; ++dt)
#pragma unroll
        for (int qt = 0; qt < 2; ++qt)
#pragma unroll
            for (int v = 0; v < 8; ++v) {
                const float x = acc[dt][qt][v];
                yv[dt][qt][v] = x * C_Y;
                pv[dt][qt][v] = x * x * C_P;
            }

#pragma unroll 1
    for (int k = 0; k < 20; ++k) {
        const float mu = -0.95f + 0.1f * (float)k;
        const float sc = KSCALE[k];
#pragma unroll
        for (int qt = 0; qt < 2; ++qt) {
            float sv = 0.0f;
#pragma unroll
            for (int dt = 0; dt < 4; ++dt)
#pragma unroll
                for (int v = 0; v < 8; ++v)
                    sv += __builtin_amdgcn_exp2f(
                        fmaf(mu, yv[dt][qt][v], pv[dt][qt][v]));
            sv += xor16(sv);
            if (!hi) atomicAdd(&s_feat[qt * 16 + l15][k], sv * sc);
        }
    }
    __syncthreads();

    // --- log1p over Ld-sums, sum over Lq --------------------------------
    if (tid < KNUM) {
        float f = 0.0f;
        for (int m = 0; m < LQ; ++m) f += log1pf(s_feat[m][tid]);
        s_f[tid] = f;
    }
    __syncthreads();

    // --- tiny MLP 21 -> 10 -> 5 -> 1 ------------------------------------
    if (tid < 10) {
        float h = b0[tid];
        for (int k = 0; k < KNUM; ++k) h += w0[tid * KNUM + k] * s_f[k];
        s_h0[tid] = fmaxf(h, 0.0f);
    }
    __syncthreads();
    if (tid < 5) {
        float h = b1[tid];
        for (int k = 0; k < 10; ++k) h += w1[tid * 10 + k] * s_h0[k];
        s_h1[tid] = fmaxf(h, 0.0f);
    }
    __syncthreads();
    if (tid == 0) {
        float h = b2[0];
        for (int k = 0; k < 5; ++k) h += w2[k] * s_h1[k];
        logits[pair * BATCH + b] = h;
    }
}

// ---------------------------------------------------------------------------
// Kernel 3: out = sigmoid(logits1 - logits2)
// ---------------------------------------------------------------------------
__global__ void sigmoid_diff_kernel(const float* __restrict__ logits,
                                    float* __restrict__ out) {
    const int i = blockIdx.x * blockDim.x + threadIdx.x;
    if (i < BATCH) {
        const float d = logits[i] - logits[BATCH + i];
        out[i] = 1.0f / (1.0f + __expf(-d));
    }
}

// ---------------------------------------------------------------------------
extern "C" void kernel_launch(void* const* d_in, const int* in_sizes, int n_in,
                              void* d_out, int out_size, void* d_ws, size_t ws_size,
                              hipStream_t stream) {
    const float* emb = (const float*)d_in[0];
    const float* w0  = (const float*)d_in[1];
    const float* b0  = (const float*)d_in[2];
    const float* w1  = (const float*)d_in[3];
    const float* b1  = (const float*)d_in[4];
    const float* w2  = (const float*)d_in[5];
    const float* b2  = (const float*)d_in[6];
    const int*   q1  = (const int*)d_in[7];
    const int*   d1  = (const int*)d_in[8];
    const int*   q2  = (const int*)d_in[9];
    const int*   d2  = (const int*)d_in[10];

    // workspace: [ f16 normalized table 50000x128 | logits 2x1024 f32 ]
    _Float16* nrm    = (_Float16*)d_ws;
    float*    logits = (float*)((char*)d_ws + (size_t)VOCAB * DIM * sizeof(_Float16));

    normalize_emb_kernel<<<(VOCAB + 7) / 8, 256, 0, stream>>>(emb, nrm);

    dim3 grid(BATCH, 2);
    knrm_kernel<<<grid, 256, 0, stream>>>(nrm, q1, d1, q2, d2,
                                          w0, b0, w1, b1, w2, b2, logits);

    sigmoid_diff_kernel<<<(BATCH + 255) / 256, 256, 0, stream>>>(logits, (float*)d_out);
}
